// BiLSTM_CRF_4844723110144
// MI455X (gfx1250) — matrix operand order repaired and checked by reading, not compiled, not run on hardware
//
#include <hip/hip_runtime.h>
#include <hip/hip_bf16.h>

typedef __attribute__((ext_vector_type(16))) __bf16 v16bf;
typedef __attribute__((ext_vector_type(8)))  float  v8f;

#define S_LEN   2048
#define LW      20
#define D_EMB   300
#define CEDIM   30
#define CO      100
#define HDIM    512
#define INDIM   400
#define KPAD    416
#define NKT     13          // 416 / 32 K-tiles
#define G4H     2048
#define NMT     128         // 2048 / 16 M-tiles
#define NNT     128         // 2048 / 16 N-tiles
#define NT      19
#define START_TAG 17
#define STOP_TAG  18
#define NEGVAL  (-10000.0f)

__device__ __forceinline__ float sigm(float x) { return 1.0f / (1.0f + expf(-x)); }

// ---------------- char CNN: conv(3,CE) pad(2,0) + maxpool over time ----------------
__global__ __launch_bounds__(128) void char_cnn_kernel(
    const int* __restrict__ chars, const float* __restrict__ char_emb,
    const float* __restrict__ conv_w, const float* __restrict__ conv_b,
    float* __restrict__ char_feat)
{
    __shared__ float ce_s[LW * CEDIM];
    const int s = blockIdx.x;
    for (int i = threadIdx.x; i < LW * CEDIM; i += 128) {
        int r = i / CEDIM, e = i % CEDIM;
        ce_s[i] = char_emb[chars[s * LW + r] * CEDIM + e];
    }
    __syncthreads();
    int co = threadIdx.x;
    if (co < CO) {
        float mx = -3.0e38f;
        for (int t = 0; t < LW + 2; ++t) {
            float acc = 0.f;
            #pragma unroll
            for (int r = 0; r < 3; ++r) {
                int row = t - 2 + r;
                if (row >= 0 && row < LW) {
                    const float* wr = conv_w + co * 3 * CEDIM + r * CEDIM;
                    const float* cr = ce_s + row * CEDIM;
                    #pragma unroll
                    for (int e = 0; e < CEDIM; ++e) acc += cr[e] * wr[e];
                }
            }
            mx = fmaxf(mx, acc);
        }
        char_feat[s * CO + co] = mx + conv_b[co];
    }
}

// ---------------- build A (emb) directly in WMMA A-fragment layout ----------------
// tile (mt, kt): 32 lanes x 16 bf16, linear idx = ((mt*NKT + kt)*32 + lane)*16 + e
// lane: row m = mt*16 + (lane&15), kbase = (lane>=16)?8:0
// e: j = e>>1; K = kt*32 + kbase + (j<4 ? 2j : 16+2(j-4)) + (e&1)
__global__ __launch_bounds__(256) void swizzle_emb_kernel(
    const int* __restrict__ sentence, const float* __restrict__ word_emb,
    const float* __restrict__ char_feat, __bf16* __restrict__ Aswz)
{
    int idx = blockIdx.x * 256 + threadIdx.x;
    if (idx >= NMT * NKT * 32 * 16) return;
    int e    = idx & 15;
    int lane = (idx >> 4) & 31;
    int tile = idx >> 9;
    int kt   = tile % NKT;
    int mt   = tile / NKT;
    int m     = mt * 16 + (lane & 15);
    int kbase = (lane >> 4) << 3;
    int j     = e >> 1;
    int k     = kt * 32 + kbase + ((j < 4) ? (2 * j) : (16 + 2 * (j - 4))) + (e & 1);
    float v = 0.f;
    if (k < D_EMB)      v = word_emb[(size_t)sentence[m] * D_EMB + k];
    else if (k < INDIM) v = char_feat[m * CO + (k - D_EMB)];
    Aswz[idx] = (__bf16)v;
}

// ---------------- build Wih^T directly in WMMA B-fragment layout + bias ----------------
// tile (kt, nt): linear idx = ((kt*NNT + nt)*32 + lane)*16 + e
// lane: col n = nt*16 + (lane&15), kbase = (lane>=16)?8:0
// e: j = e>>1; K = kt*32 + kbase + (j<4 ? 2j : 16+2(j-4)) + (e&1)
__global__ __launch_bounds__(256) void swizzle_wih_kernel(
    const float* __restrict__ wih, const float* __restrict__ bih,
    const float* __restrict__ bhh, __bf16* __restrict__ Bswz,
    float* __restrict__ biasv)
{
    int idx = blockIdx.x * 256 + threadIdx.x;
    if (idx < NKT * NNT * 32 * 16) {
        int e    = idx & 15;
        int lane = (idx >> 4) & 31;
        int tile = idx >> 9;
        int nt   = tile % NNT;
        int kt   = tile / NNT;
        int n     = nt * 16 + (lane & 15);
        int kbase = (lane >> 4) << 3;
        int k     = kt * 32 + kbase + (((e >> 1) < 4) ? (2 * (e >> 1)) : (16 + 2 * ((e >> 1) - 4))) + (e & 1);
        Bswz[idx] = (k < INDIM) ? (__bf16)wih[(size_t)n * INDIM + k] : (__bf16)0.f;
    }
    if (idx < G4H) biasv[idx] = bih[idx] + bhh[idx];
}

// ---------------- Whh -> fp32 transposed [H, 4H] ----------------
__global__ __launch_bounds__(256) void prep_whhT_kernel(
    const float* __restrict__ whh, float* __restrict__ whhT)
{
    int idx = blockIdx.x * 256 + threadIdx.x;
    if (idx >= HDIM * G4H) return;
    int j = idx % G4H, k = idx / G4H;
    whhT[(size_t)k * G4H + j] = whh[(size_t)j * HDIM + k];
}

// ---------------- WMMA GEMM on pre-swizzled fragments: xg = emb x Wih^T + bias ----------------
// block: 256 threads = 8 waves; wave -> one M-tile (16 rows) x 64 cols (4 wmma); no LDS, no barriers
__global__ __launch_bounds__(256) void gemm_xg_kernel(
    const __bf16* __restrict__ Aswz,
    const __bf16* __restrict__ Bf, const __bf16* __restrict__ Bb,
    const float* __restrict__ biasf, const float* __restrict__ biasb,
    float* __restrict__ xgf, float* __restrict__ xgb)
{
    const int dir = blockIdx.z;
    const __bf16* B   = dir ? Bb : Bf;
    const float* bias = dir ? biasb : biasf;
    float* C          = dir ? xgb : xgf;

    const int wave = threadIdx.x >> 5;
    const int lane = threadIdx.x & 31;
    const int mt   = blockIdx.y * 8 + wave;   // M-tile 0..127
    const int nt0  = blockIdx.x * 4;          // first of 4 N-tiles

    v8f acc[4] = {v8f{}, v8f{}, v8f{}, v8f{}};

    const __bf16* aptr = Aswz + (((size_t)mt * NKT) * 32 + lane) * 16;
    const __bf16* bptr = B + (((size_t)nt0 * 32) + lane) * 16;

    for (int kt = 0; kt < NKT; ++kt) {
        v16bf a = *(const v16bf*)(aptr + (size_t)kt * 512);
        const __bf16* bk = bptr + (size_t)kt * (NNT * 512);
        #pragma unroll
        for (int nn = 0; nn < 4; ++nn) {
            v16bf b = *(const v16bf*)(bk + (size_t)nn * 512);
            acc[nn] = __builtin_amdgcn_wmma_f32_16x16x32_bf16(
                false, a, false, b, (short)0, acc[nn], false, false);
        }
    }

    // C layout: VGPR v -> row = v + (lane>=16 ? 8 : 0), col = lane & 15
    const int colL  = lane & 15;
    const int rbase = mt * 16 + ((lane >> 4) << 3);
    #pragma unroll
    for (int nn = 0; nn < 4; ++nn) {
        int col = (nt0 + nn) * 16 + colL;
        float bv = bias[col];
        #pragma unroll
        for (int v = 0; v < 8; ++v) {
            C[(size_t)(rbase + v) * G4H + col] = acc[nn][v] + bv;
        }
    }
}

// ---------------- serial LSTM recurrence (one block per direction, thread = cell) ----------------
__global__ __launch_bounds__(512) void lstm_kernel(
    const float* __restrict__ xgf, const float* __restrict__ xgb,
    const float* __restrict__ whhTf, const float* __restrict__ whhTb,
    float* __restrict__ hf, float* __restrict__ hb)
{
    const int dir = blockIdx.x;
    const float* xg   = dir ? xgb : xgf;
    const float* whhT = dir ? whhTb : whhTf;
    float* hout       = dir ? hb : hf;
    const int t = threadIdx.x;           // cell index 0..511

    __shared__ float hlds[HDIM];
    hlds[t] = 0.f;
    float c = 0.f;
    __syncthreads();

    for (int step = 0; step < S_LEN; ++step) {
        int s = dir ? (S_LEN - 1 - step) : step;
        const float* xr = xg + (size_t)s * G4H;
        float ai = xr[t], af = xr[HDIM + t], ag = xr[2 * HDIM + t], ao = xr[3 * HDIM + t];
        if (step + 1 < S_LEN) {   // hide next step's gate-row fetch behind the matvec
            int sn = dir ? (S_LEN - 2 - step) : (step + 1);
            __builtin_prefetch(xg + (size_t)sn * G4H + t, 0, 3);
        }
        #pragma unroll 4
        for (int k = 0; k < HDIM; ++k) {
            float hk = hlds[k];
            const float* wr = whhT + (size_t)k * G4H;
            ai += hk * wr[t];
            af += hk * wr[HDIM + t];
            ag += hk * wr[2 * HDIM + t];
            ao += hk * wr[3 * HDIM + t];
        }
        c = sigm(af) * c + sigm(ai) * tanhf(ag);
        float h = sigm(ao) * tanhf(c);
        __syncthreads();
        hlds[t] = h;
        hout[(size_t)s * HDIM + t] = h;
        __syncthreads();
    }
}

// ---------------- tag projection: feats = [hf|hb] @ w_tag^T + b_tag ----------------
__global__ __launch_bounds__(256) void tag_proj_kernel(
    const float* __restrict__ hf, const float* __restrict__ hb,
    const float* __restrict__ w_tag, const float* __restrict__ b_tag,
    float* __restrict__ feats)
{
    int idx = blockIdx.x * 256 + threadIdx.x;
    if (idx >= S_LEN * NT) return;
    int s = idx / NT, t = idx % NT;
    float acc = b_tag[t];
    const float* wf  = w_tag + (size_t)t * (2 * HDIM);
    const float* hfp = hf + (size_t)s * HDIM;
    const float* hbp = hb + (size_t)s * HDIM;
    #pragma unroll 4
    for (int k = 0; k < HDIM; ++k) acc += hfp[k] * wf[k];
    #pragma unroll 4
    for (int k = 0; k < HDIM; ++k) acc += hbp[k] * wf[HDIM + k];
    feats[idx] = acc;
}

// ---------------- Viterbi decode (single wave32 block) ----------------
__global__ __launch_bounds__(32) void viterbi_kernel(
    const float* __restrict__ feats, const float* __restrict__ transitions,
    int* __restrict__ bp, float* __restrict__ out)
{
    __shared__ float tr[NT * NT];
    __shared__ float fv[32];
    __shared__ float nfv[32];
    const int t = threadIdx.x;
    for (int i = t; i < NT * NT; i += 32) tr[i] = transitions[i];
    fv[t] = (t == START_TAG) ? 0.f : NEGVAL;
    __syncthreads();

    for (int s = 0; s < S_LEN; ++s) {
        if (t < NT) {
            float best = -3.0e38f; int bi = 0;
            #pragma unroll
            for (int p = 0; p < NT; ++p) {
                float v = fv[p] + tr[t * NT + p];
                if (v > best) { best = v; bi = p; }
            }
            bp[s * NT + t] = bi;
            nfv[t] = best + feats[s * NT + t];
        }
        __syncthreads();
        if (t < NT) fv[t] = nfv[t];
        __syncthreads();
    }

    if (t == 0) {
        float bscore = -3.0e38f; int best = 0;
        for (int p = 0; p < NT; ++p) {
            float v = fv[p] + tr[STOP_TAG * NT + p];
            if (v > bscore) { bscore = v; best = p; }
        }
        out[0] = bscore;
        int cur = best;
        out[1 + (S_LEN - 1)] = (float)best;
        for (int s = S_LEN - 1; s >= 1; --s) {
            cur = bp[s * NT + cur];
            out[1 + (s - 1)] = (float)cur;
        }
    }
}

extern "C" void kernel_launch(void* const* d_in, const int* in_sizes, int n_in,
                              void* d_out, int out_size, void* d_ws, size_t ws_size,
                              hipStream_t stream) {
    const int*   sentence = (const int*)d_in[0];
    const int*   chars    = (const int*)d_in[1];
    const float* word_emb = (const float*)d_in[4];
    const float* char_emb = (const float*)d_in[5];
    const float* conv_w   = (const float*)d_in[6];
    const float* conv_b   = (const float*)d_in[7];
    const float* wih_f    = (const float*)d_in[8];
    const float* whh_f    = (const float*)d_in[9];
    const float* bih_f    = (const float*)d_in[10];
    const float* bhh_f    = (const float*)d_in[11];
    const float* wih_b    = (const float*)d_in[12];
    const float* whh_b    = (const float*)d_in[13];
    const float* bih_b    = (const float*)d_in[14];
    const float* bhh_b    = (const float*)d_in[15];
    const float* w_tag    = (const float*)d_in[16];
    const float* b_tag    = (const float*)d_in[17];
    const float* trans    = (const float*)d_in[18];

    char* ws = (char*)d_ws;
    size_t off = 0;
    auto take = [&](size_t bytes) -> char* {
        char* p = ws + off; off += (bytes + 255) & ~(size_t)255; return p;
    };
    float*  char_feat = (float*) take((size_t)S_LEN * CO * 4);
    __bf16* Aswz      = (__bf16*)take((size_t)NMT * NKT * 512 * 2);
    __bf16* Bswz_f    = (__bf16*)take((size_t)NKT * NNT * 512 * 2);
    __bf16* Bswz_b    = (__bf16*)take((size_t)NKT * NNT * 512 * 2);
    float*  biasv_f   = (float*) take((size_t)G4H * 4);
    float*  biasv_b   = (float*) take((size_t)G4H * 4);
    float*  xg_f      = (float*) take((size_t)S_LEN * G4H * 4);
    float*  xg_b      = (float*) take((size_t)S_LEN * G4H * 4);
    float*  whhT_f    = (float*) take((size_t)HDIM * G4H * 4);
    float*  whhT_b    = (float*) take((size_t)HDIM * G4H * 4);
    float*  hf        = (float*) take((size_t)S_LEN * HDIM * 4);
    float*  hb        = (float*) take((size_t)S_LEN * HDIM * 4);
    float*  feats     = (float*) take((size_t)S_LEN * NT * 4);
    int*    bp        = (int*)   take((size_t)S_LEN * NT * 4);

    const int frag_elems = NMT * NKT * 512;  // == NKT * NNT * 512

    // 1. char CNN
    char_cnn_kernel<<<S_LEN, 128, 0, stream>>>(chars, char_emb, conv_w, conv_b, char_feat);
    // 2. A operand in fragment layout
    swizzle_emb_kernel<<<(frag_elems + 255) / 256, 256, 0, stream>>>(sentence, word_emb, char_feat, Aswz);
    // 3. weight prep (B operands in fragment layout, Whh transposed fp32)
    swizzle_wih_kernel<<<(frag_elems + 255) / 256, 256, 0, stream>>>(wih_f, bih_f, bhh_f, Bswz_f, biasv_f);
    swizzle_wih_kernel<<<(frag_elems + 255) / 256, 256, 0, stream>>>(wih_b, bih_b, bhh_b, Bswz_b, biasv_b);
    prep_whhT_kernel<<<(HDIM * G4H + 255) / 256, 256, 0, stream>>>(whh_f, whhT_f);
    prep_whhT_kernel<<<(HDIM * G4H + 255) / 256, 256, 0, stream>>>(whh_b, whhT_b);
    // 4. WMMA input projections (both directions)
    gemm_xg_kernel<<<dim3(NNT / 4, NMT / 8, 2), 256, 0, stream>>>(
        Aswz, Bswz_f, Bswz_b, biasv_f, biasv_b, xg_f, xg_b);
    // 5. serial recurrences
    lstm_kernel<<<2, HDIM, 0, stream>>>(xg_f, xg_b, whhT_f, whhT_b, hf, hb);
    // 6. tag projection
    tag_proj_kernel<<<(S_LEN * NT + 255) / 256, 256, 0, stream>>>(hf, hb, w_tag, b_tag, feats);
    // 7. Viterbi + backtrack -> d_out
    viterbi_kernel<<<1, 32, 0, stream>>>(feats, trans, bp, (float*)d_out);
}
